// MMTransformer_89799176224933
// MI455X (gfx1250) — matrix operand order repaired and checked by reading, not compiled, not run on hardware
//
#include <hip/hip_runtime.h>
#include <math.h>

// ---------------- model constants ----------------
constexpr int CB   = 32;
constexpr int CIMG = 49;
constexpr int CTXT = 77;
constexpr int CD   = 512;
constexpr int CH   = 8;
constexpr int CHS  = 64;
constexpr int CNB  = 4;
constexpr int CFF  = 2048;
constexpr int CV   = 49408;
constexpr int CL   = CIMG + CTXT;   // 126

// ---------------- WMMA / vector types ----------------
typedef __attribute__((ext_vector_type(16))) __bf16        v16bf;
typedef __attribute__((ext_vector_type(2)))  __bf16        v2bf;
typedef __attribute__((ext_vector_type(8)))  float         v8f;
typedef __attribute__((ext_vector_type(4)))  unsigned int  u32x4;
typedef __attribute__((ext_vector_type(2)))  unsigned int  u32x2;
typedef __attribute__((ext_vector_type(8)))  int           i32x8;
typedef __attribute__((ext_vector_type(4)))  int           i32x4;

union B16Frag { u32x4 q[2]; v16bf v; };

__device__ __forceinline__ unsigned short f2bf(float f) {
    unsigned int u = __float_as_uint(f);
    u += 0x7FFFu + ((u >> 16) & 1u);          // round-to-nearest-even
    return (unsigned short)(u >> 16);
}

#if defined(__has_builtin)
#if __has_builtin(__builtin_amdgcn_cvt_pk_bf16_f32)
#define HAVE_CVT_PK_BF16 1
#endif
#endif

__device__ __forceinline__ unsigned int pack2(float a, float b) {
#ifdef HAVE_CVT_PK_BF16
    union { v2bf v; unsigned int u; } cv;
    cv.v = __builtin_amdgcn_cvt_pk_bf16_f32(a, b);
    return cv.u;
#else
    return (unsigned int)f2bf(a) | ((unsigned int)f2bf(b) << 16);
#endif
}
__device__ __forceinline__ float gelu_exact(float v) {
    return 0.5f * v * (1.0f + erff(v * 0.70710678118654752f));
}

// =====================================================================
// Weight conversion: fp32 [K][N] (or QKV [h][d][e]) -> bf16 transposed [N][K]
// =====================================================================
__global__ __launch_bounds__(256)
void cvt_w(const float* __restrict__ src, unsigned short* __restrict__ dst,
           int K, int N, int qkv)
{
    long long idx = (long long)blockIdx.x * 256 + threadIdx.x;
    if (idx >= (long long)K * N) return;
    int k = (int)(idx % K);
    int n = (int)(idx / K);
    size_t off = qkv ? (((size_t)(n >> 6) * K + k) * 64 + (n & 63))
                     : ((size_t)k * N + n);
    dst[idx] = f2bf(src[off]);
}

// =====================================================================
// TDM: async DMA of one 128-row x 32-elem bf16 tile (row stride K elems)
// into LDS with 16-byte padding every 64-byte row (-> 40-ushort pitch).
// Issued by one wave; completion via s_wait_tensorcnt.
// =====================================================================
__device__ __forceinline__ void tdm_load_b_tile(const unsigned short* gptr,
                                                unsigned lds_off, int K)
{
    unsigned long long ga = (unsigned long long)(const void*)gptr;
    u32x4 g0;
    g0[0] = 1u;                                            // count=1, user mode
    g0[1] = lds_off;                                       // lds_addr (bytes)
    g0[2] = (unsigned)ga;                                  // global_addr[31:0]
    g0[3] = (unsigned)((ga >> 32) & 0x1FFFFFFu)            // global_addr[56:32]
          | (2u << 30);                                    // type=2 ("image")
    i32x8 g1;
    g1[0] = (1 << 16)                                      // data_size = 2 bytes
          | (1 << 20)                                      // pad_enable
          | (3 << 22)                                      // pad_interval: 16 DWORDs (64B row)
          | (3 << 25);                                     // pad_amount: 4 DWORDs (16B)
    g1[1] = (int)(32u << 16);                              // tensor_dim0 = 32 (lo16)
    g1[2] = (int)(128u << 16);                             // tensor_dim1 = 128 (lo16)
    g1[3] = (int)(32u << 16);                              // tile_dim0 = 32
    g1[4] = 128;                                           // tile_dim1 = 128
    g1[5] = K;                                             // tensor_dim0_stride lo32
    g1[6] = 0;
    g1[7] = 0;
    i32x4 z4; z4[0] = 0; z4[1] = 0; z4[2] = 0; z4[3] = 0;
    i32x8 z8;
    #pragma unroll
    for (int i = 0; i < 8; ++i) z8[i] = 0;
    __builtin_amdgcn_tensor_load_to_lds(g0, g1, z4, z4, z8, 0);
}

// =====================================================================
// Tiled GEMM: C[M,N] = act( A_f32[M,K] @ Bt_bf16[N][K]^T + bias + resid )
// bf16 WMMA 16x16x32, fp32 accumulate; double-buffered LDS.
// B tile moved by the Tensor Data Mover; A converted fp32->bf16 in VGPRs.
// Requires: K % 64 == 0 (nk even), N % 128 == 0.
// =====================================================================
#define GBM 64
#define GBN 128
#define GBK 32
#define GLDA (GBK + 8)   // 40 ushorts
#define GLDB (GBK + 8)   // matches TDM pad: 64B row + 16B pad

__global__ __launch_bounds__(256)
void gemm_bf16_wmma(const float* __restrict__ A, const unsigned short* __restrict__ Bt,
                    const float* __restrict__ bias, const float* __restrict__ resid,
                    float* __restrict__ C, int M, int N, int K, int act)
{
    __shared__ __align__(16) unsigned short sA[2][GBM * GLDA];
    __shared__ __align__(16) unsigned short sB[2][GBN * GLDB];

    const int tid  = threadIdx.x;
    const int lane = tid & 31;
    const int wave = tid >> 5;
    const int wm   = wave & 1;
    const int wn   = wave >> 1;
    const int bm0  = blockIdx.y * GBM;
    const int bn0  = blockIdx.x * GBN;
    const int hi   = lane >> 4;
    const int l16  = lane & 15;

    // A staging coords: two float4 pieces per thread
    int am[2], ak[2];
    const float* aptr[2];
    #pragma unroll
    for (int it = 0; it < 2; ++it) {
        int idx = tid + it * 256;
        am[it] = idx >> 3;
        ak[it] = (idx & 7) << 2;
        int gm = bm0 + am[it];
        if (gm > M - 1) gm = M - 1;   // clamp: OOB rows recompute row M-1, never stored
        aptr[it] = A + (size_t)gm * K + ak[it];
    }

    const unsigned short* btile = Bt + (size_t)bn0 * K;
    const unsigned ldsB0 = (unsigned)(unsigned long long)(const void*)&sB[0][0];
    const unsigned ldsB1 = (unsigned)(unsigned long long)(const void*)&sB[1][0];

    v8f acc[2][2];
    #pragma unroll
    for (int i = 0; i < 2; ++i)
        #pragma unroll
        for (int j = 0; j < 2; ++j)
            #pragma unroll
            for (int e = 0; e < 8; ++e) acc[i][j][e] = 0.0f;

    float4 va[2];

#define STAGE_A(buf)                                                      \
    do {                                                                  \
        _Pragma("unroll")                                                 \
        for (int it = 0; it < 2; ++it) {                                  \
            u32x2 w;                                                      \
            w.x = pack2(va[it].x, va[it].y);                              \
            w.y = pack2(va[it].z, va[it].w);                              \
            *(u32x2*)&sA[buf][am[it] * GLDA + ak[it]] = w;                \
        }                                                                 \
    } while (0)

#define COMPUTE(cur)                                                      \
    do {                                                                  \
        B16Frag af[2], bf[2];                                             \
        _Pragma("unroll")                                                 \
        for (int i = 0; i < 2; ++i) {                                     \
            int row = wm * 32 + i * 16 + l16;                             \
            const u32x4* p = (const u32x4*)&sA[cur][row * GLDA + hi * 8]; \
            af[i].q[0] = p[0]; af[i].q[1] = p[2];                         \
        }                                                                 \
        _Pragma("unroll")                                                 \
        for (int j = 0; j < 2; ++j) {                                     \
            int n = wn * 32 + j * 16 + l16;                               \
            const u32x4* p = (const u32x4*)&sB[cur][n * GLDB + hi * 8];   \
            bf[j].q[0] = p[0]; bf[j].q[1] = p[2];                         \
        }                                                                 \
        _Pragma("unroll")                                                 \
        for (int i = 0; i < 2; ++i)                                       \
            _Pragma("unroll")                                             \
            for (int j = 0; j < 2; ++j)                                   \
                acc[i][j] = __builtin_amdgcn_wmma_f32_16x16x32_bf16(      \
                    false, af[i].v, false, bf[j].v, (short)0,             \
                    acc[i][j], false, false);                             \
    } while (0)

    // prologue: stage K-tile 0 (TDM for B, VALU-converted A)
    if (wave == 0) tdm_load_b_tile(btile, ldsB0, K);
    #pragma unroll
    for (int it = 0; it < 2; ++it) va[it] = *(const float4*)(aptr[it]);
    STAGE_A(0);
    if (wave == 0) __builtin_amdgcn_s_wait_tensorcnt(0);
    __syncthreads();

    const int nk = K >> 5;                 // even for all K used here
    for (int kt = 0; kt < nk; kt += 2) {
        // ---- step A: consume buf 0, prefetch into buf 1 ----
        {
            const int kn = (kt + 1) << 5;  // kt+1 < nk always (nk even)
            if (wave == 0) tdm_load_b_tile(btile + kn, ldsB1, K);
            #pragma unroll
            for (int it = 0; it < 2; ++it) va[it] = *(const float4*)(aptr[it] + kn);
        }
        COMPUTE(0);
        STAGE_A(1);
        if (wave == 0) __builtin_amdgcn_s_wait_tensorcnt(0);
        __syncthreads();

        // ---- step B: consume buf 1, prefetch into buf 0 ----
        const bool more = (kt + 2 < nk);
        if (more) {
            const int kn = (kt + 2) << 5;
            if (wave == 0) tdm_load_b_tile(btile + kn, ldsB0, K);
            #pragma unroll
            for (int it = 0; it < 2; ++it) va[it] = *(const float4*)(aptr[it] + kn);
        }
        COMPUTE(1);
        if (more) STAGE_A(0);
        if (wave == 0) __builtin_amdgcn_s_wait_tensorcnt(0);
        __syncthreads();
    }
#undef STAGE_A
#undef COMPUTE

    // epilogue: C layout — lanes 0-15: M=r, lanes 16-31: M=8+r; N=l16
    #pragma unroll
    for (int i = 0; i < 2; ++i) {
        #pragma unroll
        for (int j = 0; j < 2; ++j) {
            int gn = bn0 + wn * 32 + j * 16 + l16;
            float bsv = bias ? bias[gn] : 0.0f;
            #pragma unroll
            for (int r = 0; r < 8; ++r) {
                int gm = bm0 + wm * 32 + i * 16 + r + hi * 8;
                if (gm < M) {
                    float v = acc[i][j][r] + bsv;
                    if (resid) v += resid[(size_t)gm * N + gn];
                    if (act == 1) v = gelu_exact(v);
                    C[(size_t)gm * N + gn] = v;
                }
            }
        }
    }
}

// =====================================================================
// Attention scores: per (b,h), S[l,m] = scale * Q·Kᵀ with block mask
// =====================================================================
#define SLD (64 + 8)
__global__ __launch_bounds__(256)
void attn_scores(const float* __restrict__ qb, const float* __restrict__ kb,
                 const int* __restrict__ amask, float* __restrict__ S, float scale)
{
    const int bh = blockIdx.x;
    const int b  = bh / CH;
    const int h  = bh % CH;
    __shared__ __align__(16) unsigned short sQ[128 * SLD];
    __shared__ __align__(16) unsigned short sK[128 * SLD];
    const int tid = threadIdx.x;

    #pragma unroll
    for (int it = 0; it < 8; ++it) {
        int idx = tid + it * 256;
        int l  = idx >> 4;
        int e4 = (idx & 15) << 2;
        int lc = (l > CL - 1) ? CL - 1 : l;
        size_t off = ((size_t)(b * CL + lc)) * CD + h * CHS + e4;
        float4 q4 = *(const float4*)(qb + off);
        float4 k4 = *(const float4*)(kb + off);
        u32x2 wq; wq.x = pack2(q4.x, q4.y); wq.y = pack2(q4.z, q4.w);
        u32x2 wk; wk.x = pack2(k4.x, k4.y); wk.y = pack2(k4.z, k4.w);
        *(u32x2*)&sQ[l * SLD + e4] = wq;
        *(u32x2*)&sK[l * SLD + e4] = wk;
    }
    __syncthreads();

    const int lane = tid & 31, wave = tid >> 5;
    const int hi = lane >> 4, l16 = lane & 15;
    const int ti = wave;

    B16Frag af[2];
    #pragma unroll
    for (int ks = 0; ks < 2; ++ks) {
        int row = ti * 16 + l16;
        const u32x4* p = (const u32x4*)&sQ[row * SLD + ks * 32 + hi * 8];
        af[ks].q[0] = p[0];
        af[ks].q[1] = p[2];
    }

    for (int tj = 0; tj < 8; ++tj) {
        v8f acc;
        for (int e = 0; e < 8; ++e) acc[e] = 0.0f;
        #pragma unroll
        for (int ks = 0; ks < 2; ++ks) {
            B16Frag bfr;
            int n = tj * 16 + l16;
            const u32x4* p = (const u32x4*)&sK[n * SLD + ks * 32 + hi * 8];
            bfr.q[0] = p[0]; bfr.q[1] = p[2];
            acc = __builtin_amdgcn_wmma_f32_16x16x32_bf16(
                false, af[ks].v, false, bfr.v, (short)0, acc, false, false);
        }
        int km = tj * 16 + l16;
        bool kpad = (km < CIMG) ? true
                  : (km < CL ? (amask[b * CTXT + km - CIMG] != 0) : false);
        #pragma unroll
        for (int r = 0; r < 8; ++r) {
            int gl = ti * 16 + r + hi * 8;
            if (gl < CL && km < CL) {
                bool allowed = kpad && !((gl >= CIMG) && (km >= CIMG) && (km > gl));
                float v = allowed ? acc[r] * scale : -1e30f;
                S[((size_t)bh * 128 + gl) * 128 + km] = v;
            }
        }
    }
}

// =====================================================================
// Softmax over 126 valid cols of each padded 128-col row; zero padding
// =====================================================================
__global__ __launch_bounds__(128)
void softmax_rows(float* __restrict__ S)
{
    const int row = blockIdx.x;
    const int bh  = row / CL;
    const int l   = row % CL;
    float* p = S + ((size_t)bh * 128 + l) * 128;
    const int t = threadIdx.x;
    __shared__ float red[128];

    float v = (t < CL) ? p[t] : -1e30f;
    red[t] = v; __syncthreads();
    for (int s = 64; s > 0; s >>= 1) { if (t < s) red[t] = fmaxf(red[t], red[t + s]); __syncthreads(); }
    float mx = red[0]; __syncthreads();

    float e = (t < CL) ? __expf(v - mx) : 0.0f;
    red[t] = e; __syncthreads();
    for (int s = 64; s > 0; s >>= 1) { if (t < s) red[t] += red[t + s]; __syncthreads(); }
    float sum = red[0];

    p[t] = e / sum;
}

// =====================================================================
// O = W @ V per (b,h): [126x126] @ [126x64], K padded to 128
// =====================================================================
#define WLD (128 + 8)
__global__ __launch_bounds__(256)
void attn_av(const float* __restrict__ S, const float* __restrict__ vbuf,
             float* __restrict__ O)
{
    const int bh = blockIdx.x;
    const int b  = bh / CH;
    const int h  = bh % CH;
    __shared__ __align__(16) unsigned short sW[128 * WLD];
    __shared__ __align__(16) unsigned short sV[64 * WLD];
    const int tid = threadIdx.x;

    #pragma unroll
    for (int it = 0; it < 16; ++it) {
        int idx = tid + it * 256;
        int l  = idx >> 5;
        int m4 = (idx & 31) << 2;
        int lc = (l > CL - 1) ? CL - 1 : l;
        float4 w4 = *(const float4*)(S + ((size_t)bh * 128 + lc) * 128 + m4);
        u32x2 w; w.x = pack2(w4.x, w4.y); w.y = pack2(w4.z, w4.w);
        *(u32x2*)&sW[l * WLD + m4] = w;
    }
    #pragma unroll
    for (int it = 0; it < 8; ++it) {
        int idx = tid + it * 256;
        int m  = idx >> 4;
        int e4 = (idx & 15) << 2;
        int mc = (m > CL - 1) ? CL - 1 : m;
        float z = (m < CL) ? 1.0f : 0.0f;
        float4 v4 = *(const float4*)(vbuf + ((size_t)(b * CL + mc)) * CD + h * CHS + e4);
        sV[(e4 + 0) * WLD + m] = f2bf(v4.x * z);
        sV[(e4 + 1) * WLD + m] = f2bf(v4.y * z);
        sV[(e4 + 2) * WLD + m] = f2bf(v4.z * z);
        sV[(e4 + 3) * WLD + m] = f2bf(v4.w * z);
    }
    __syncthreads();

    const int lane = tid & 31, wave = tid >> 5;
    const int hi = lane >> 4, l16 = lane & 15;
    const int ti = wave;

    for (int tj = 0; tj < 4; ++tj) {
        v8f acc;
        for (int e = 0; e < 8; ++e) acc[e] = 0.0f;
        #pragma unroll
        for (int kt = 0; kt < 4; ++kt) {
            B16Frag afr, bfr;
            int row = ti * 16 + l16;
            const u32x4* pa = (const u32x4*)&sW[row * WLD + kt * 32 + hi * 8];
            afr.q[0] = pa[0]; afr.q[1] = pa[2];
            int e = tj * 16 + l16;
            const u32x4* pb = (const u32x4*)&sV[e * WLD + kt * 32 + hi * 8];
            bfr.q[0] = pb[0]; bfr.q[1] = pb[2];
            acc = __builtin_amdgcn_wmma_f32_16x16x32_bf16(
                false, afr.v, false, bfr.v, (short)0, acc, false, false);
        }
        int ge = tj * 16 + l16;
        #pragma unroll
        for (int r = 0; r < 8; ++r) {
            int gl = ti * 16 + r + hi * 8;
            if (gl < CL)
                O[((size_t)(b * CL + gl)) * CD + h * CHS + ge] = acc[r];
        }
    }
}

// =====================================================================
// LayerNorm: one 128-thread block per row of D=512, float4 per thread.
// =====================================================================
__global__ __launch_bounds__(128)
void layernorm_k(const float* __restrict__ x, const float* __restrict__ g,
                 const float* __restrict__ be, float* __restrict__ o, int compact)
{
    int row = blockIdx.x;
    int src = compact ? ((row / CTXT) * CL + CIMG + (row % CTXT)) : row;
    const int t = threadIdx.x;
    float4 lv = ((const float4*)(x + (size_t)src * CD))[t];
    float s = lv.x + lv.y + lv.z + lv.w;
    __shared__ float red[128];
    red[t] = s; __syncthreads();
    for (int st = 64; st > 0; st >>= 1) { if (t < st) red[t] += red[t + st]; __syncthreads(); }
    float mean = red[0] * (1.0f / CD); __syncthreads();
    float dx = lv.x - mean, dy = lv.y - mean, dz = lv.z - mean, dw = lv.w - mean;
    red[t] = dx * dx + dy * dy + dz * dz + dw * dw; __syncthreads();
    for (int st = 64; st > 0; st >>= 1) { if (t < st) red[t] += red[t + st]; __syncthreads(); }
    float rstd = rsqrtf(red[0] * (1.0f / CD) + 1e-5f);
    float4 gv = ((const float4*)g)[t];
    float4 bv = ((const float4*)be)[t];
    float4 ov;
    ov.x = dx * rstd * gv.x + bv.x;
    ov.y = dy * rstd * gv.y + bv.y;
    ov.z = dz * rstd * gv.z + bv.z;
    ov.w = dw * rstd * gv.w + bv.w;
    ((float4*)(o + (size_t)row * CD))[t] = ov;
}

// =====================================================================
// Assemble x[b,l,d] = (l<IMG ? img_proj : txt_embed) + pos[l,d]
// =====================================================================
__global__ __launch_bounds__(256)
void assemble_x(const float* __restrict__ img, const float* __restrict__ txt,
                const float* __restrict__ pos, float* __restrict__ x)
{
    int idx = blockIdx.x * 256 + threadIdx.x;
    if (idx >= CB * CL * CD / 4) return;
    int q  = idx & 127;
    int bl = idx >> 7;
    int l  = bl % CL;
    int b  = bl / CL;
    float4 v = (l < CIMG)
             ? ((const float4*)img)[((size_t)(b * CIMG + l)) * 128 + q]
             : ((const float4*)txt)[((size_t)(b * CTXT + (l - CIMG))) * 128 + q];
    float4 p = ((const float4*)pos)[l * 128 + q];
    float4 o; o.x = v.x + p.x; o.y = v.y + p.y; o.z = v.z + p.z; o.w = v.w + p.w;
    ((float4*)x)[idx] = o;
}

// =====================================================================
// Host-side orchestration
// =====================================================================
static inline int cdiv(int a, int b) { return (a + b - 1) / b; }

extern "C" void kernel_launch(void* const* d_in, const int* in_sizes, int n_in,
                              void* d_out, int out_size, void* d_ws, size_t ws_size,
                              hipStream_t stream)
{
    (void)in_sizes; (void)n_in; (void)out_size; (void)ws_size;
    const float* img_hidden  = (const float*)d_in[0];
    const float* txt_embed   = (const float*)d_in[1];
    const float* image_proj_w= (const float*)d_in[2];
    const float* image_proj_b= (const float*)d_in[3];
    const float* pos_embed   = (const float*)d_in[4];
    const float* Wq          = (const float*)d_in[5];
    const float* Wk          = (const float*)d_in[6];
    const float* Wv          = (const float*)d_in[7];
    const float* attn_proj_w = (const float*)d_in[8];
    const float* attn_proj_b = (const float*)d_in[9];
    const float* ln1_g       = (const float*)d_in[10];
    const float* ln1_b       = (const float*)d_in[11];
    const float* ln2_g       = (const float*)d_in[12];
    const float* ln2_b       = (const float*)d_in[13];
    const float* mlp_w1      = (const float*)d_in[14];
    const float* mlp_b1      = (const float*)d_in[15];
    const float* mlp_w2      = (const float*)d_in[16];
    const float* mlp_b2      = (const float*)d_in[17];
    const float* lnf_g       = (const float*)d_in[18];
    const float* lnf_b       = (const float*)d_in[19];
    const float* out_w       = (const float*)d_in[20];
    const float* out_b       = (const float*)d_in[21];
    const int*   amask       = (const int*)d_in[22];
    float* logits            = (float*)d_out;

    const int ROWS  = CB * CL;    // 4032
    const int TROWS = CB * CTXT;  // 2464

    float* ws     = (float*)d_ws;
    float* x      = ws;
    float* hbuf   = x    + (size_t)ROWS * CD;
    float* qb     = hbuf + (size_t)ROWS * CD;
    float* kb     = qb   + (size_t)ROWS * CD;
    float* vb     = kb   + (size_t)ROWS * CD;
    float* scores = vb   + (size_t)ROWS * CD;
    float* ob     = scores + (size_t)CB * CH * 128 * 128;
    float* mid    = ob   + (size_t)ROWS * CD;
    float* ximg   = mid  + (size_t)ROWS * CFF;
    float* xf     = ximg + (size_t)CB * CIMG * CD;

    unsigned short* wimg = (unsigned short*)(xf + (size_t)TROWS * CD);
    unsigned short* wblk = wimg + (size_t)CD * 768;
    const size_t SQ  = (size_t)CD * CD;
    const size_t SBL = 4 * SQ + 2 * (size_t)CD * CFF;
    unsigned short* wout = wblk + CNB * SBL;

    const float scale = 0.04419417382415922f;  // 1/sqrt(512)

    // ---------------- weight conversion ----------------
    cvt_w<<<cdiv(768 * CD, 256), 256, 0, stream>>>(image_proj_w, wimg, 768, CD, 0);
    for (int i = 0; i < CNB; ++i) {
        unsigned short* wbi = wblk + i * SBL;
        const size_t wstride = (size_t)CH * CD * CHS;
        cvt_w<<<cdiv(CD * CD, 256), 256, 0, stream>>>(Wq + i * wstride, wbi + 0 * SQ, CD, CD, 1);
        cvt_w<<<cdiv(CD * CD, 256), 256, 0, stream>>>(Wk + i * wstride, wbi + 1 * SQ, CD, CD, 1);
        cvt_w<<<cdiv(CD * CD, 256), 256, 0, stream>>>(Wv + i * wstride, wbi + 2 * SQ, CD, CD, 1);
        cvt_w<<<cdiv(CD * CD, 256), 256, 0, stream>>>(attn_proj_w + i * SQ, wbi + 3 * SQ, CD, CD, 0);
        cvt_w<<<cdiv(CD * CFF, 256), 256, 0, stream>>>(mlp_w1 + (size_t)i * CD * CFF, wbi + 4 * SQ, CD, CFF, 0);
        cvt_w<<<cdiv(CD * CFF, 256), 256, 0, stream>>>(mlp_w2 + (size_t)i * CFF * CD, wbi + 4 * SQ + (size_t)CD * CFF, CFF, CD, 0);
    }
    cvt_w<<<cdiv(CD * CV, 256), 256, 0, stream>>>(out_w, wout, CD, CV, 0);

    // ---------------- forward pass ----------------
    gemm_bf16_wmma<<<dim3(CD / GBN, cdiv(CB * CIMG, GBM)), 256, 0, stream>>>(
        img_hidden, wimg, image_proj_b, nullptr, ximg, CB * CIMG, CD, 768, 0);

    assemble_x<<<cdiv(CB * CL * CD / 4, 256), 256, 0, stream>>>(ximg, txt_embed, pos_embed, x);

    for (int i = 0; i < CNB; ++i) {
        unsigned short* wbi = wblk + i * SBL;

        layernorm_k<<<ROWS, 128, 0, stream>>>(x, ln1_g + i * CD, ln1_b + i * CD, hbuf, 0);

        gemm_bf16_wmma<<<dim3(CD / GBN, cdiv(ROWS, GBM)), 256, 0, stream>>>(
            hbuf, wbi + 0 * SQ, nullptr, nullptr, qb, ROWS, CD, CD, 0);
        gemm_bf16_wmma<<<dim3(CD / GBN, cdiv(ROWS, GBM)), 256, 0, stream>>>(
            hbuf, wbi + 1 * SQ, nullptr, nullptr, kb, ROWS, CD, CD, 0);
        gemm_bf16_wmma<<<dim3(CD / GBN, cdiv(ROWS, GBM)), 256, 0, stream>>>(
            hbuf, wbi + 2 * SQ, nullptr, nullptr, vb, ROWS, CD, CD, 0);

        attn_scores<<<CB * CH, 256, 0, stream>>>(qb, kb, amask, scores, scale);
        softmax_rows<<<CB * CH * CL, 128, 0, stream>>>(scores);
        attn_av<<<CB * CH, 256, 0, stream>>>(scores, vb, ob);

        gemm_bf16_wmma<<<dim3(CD / GBN, cdiv(ROWS, GBM)), 256, 0, stream>>>(
            ob, wbi + 3 * SQ, attn_proj_b + i * CD, x, x, ROWS, CD, CD, 0);

        layernorm_k<<<ROWS, 128, 0, stream>>>(x, ln2_g + i * CD, ln2_b + i * CD, hbuf, 0);
        gemm_bf16_wmma<<<dim3(CFF / GBN, cdiv(ROWS, GBM)), 256, 0, stream>>>(
            hbuf, wbi + 4 * SQ, mlp_b1 + i * CFF, nullptr, mid, ROWS, CFF, CD, 1);
        gemm_bf16_wmma<<<dim3(CD / GBN, cdiv(ROWS, GBM)), 256, 0, stream>>>(
            mid, wbi + 4 * SQ + (size_t)CD * CFF, mlp_b2 + i * CD, x, x, ROWS, CD, CFF, 0);
    }

    layernorm_k<<<TROWS, 128, 0, stream>>>(x, lnf_g, lnf_b, xf, 1);
    gemm_bf16_wmma<<<dim3(CV / GBN, cdiv(TROWS, GBM)), 256, 0, stream>>>(
        xf, wout, out_b, nullptr, logits, TROWS, CV, CD, 0);
}
